// MyModel_1632087573354
// MI455X (gfx1250) — compile-verified, matrix operand order checked by reading, not run on hardware
//
#include <hip/hip_runtime.h>
#include <hip/hip_bf16.h>

// ---------------------------------------------------------------------------
// CDNA5 (gfx1250) VQ + scoring pipeline.
//   score = query @ codebook[argmin_k ||ent - cb_k||]^T   (bf16 WMMA, f32 acc)
//   vq_loss = (1+beta) * mse(quant, entity)               (fp32)
// ---------------------------------------------------------------------------

typedef __attribute__((ext_vector_type(16))) __bf16 v16bf;
typedef __attribute__((ext_vector_type(8)))  float  v8f;
typedef int v4i __attribute__((vector_size(16)));

#define DIMS 512           // D
#define BETA 0.25f
#define APAD 16            // bf16 pad per LDS row -> 1056B stride (8-bank skew)

// ---------------------------------------------------------------------------
// Async global->LDS staging (gfx1250 GLOBAL_LOAD_ASYNC_TO_LDS_B128, ASYNCcnt),
// with a portable fallback. Builtin signature (from toolchain diagnostic):
//   (v4i addrspace(1)*, v4i addrspace(3)*, imm int offset, imm int cpol).
// LDS operand: per ISA, generic LDS addresses carry the LDS byte offset in the
// low 32 bits, and AS(3) pointers are 32-bit, so truncate via integer.
// ---------------------------------------------------------------------------
#if defined(__has_builtin)
#if __has_builtin(__builtin_amdgcn_global_load_async_to_lds_b128)
#define HAVE_ASYNC_LDS 1
#endif
#endif

__device__ __forceinline__ void async_copy16(void* lds_dst, const void* gsrc) {
#if defined(HAVE_ASYNC_LDS)
  typedef __attribute__((address_space(1))) v4i gv4i;
  typedef __attribute__((address_space(3))) v4i lv4i;
  __builtin_amdgcn_global_load_async_to_lds_b128(
      (gv4i*)(uintptr_t)gsrc,
      (lv4i*)(uint32_t)(uintptr_t)lds_dst,
      0, 0);
#else
  *(uint4*)lds_dst = *(const uint4*)gsrc;
#endif
}

__device__ __forceinline__ void async_copy_join() {
#if defined(HAVE_ASYNC_LDS)
  asm volatile("s_wait_asynccnt 0x0" ::: "memory");
#endif
  __syncthreads();
}

// ---------------------------------------------------------------------------
// fp32 -> bf16 row conversion, optional squared-norm output. One wave per row.
// ---------------------------------------------------------------------------
__global__ void k_convert(const float* __restrict__ src,
                          __bf16* __restrict__ dst,
                          float* __restrict__ c2,
                          int rows) {
  const int row  = blockIdx.x * 8 + (threadIdx.x >> 5);
  const int lane = threadIdx.x & 31;
  if (row >= rows) return;

  const float4* s = reinterpret_cast<const float4*>(src + (size_t)row * DIMS) + lane * 4;
  float sum = 0.0f;
  v16bf t;
#pragma unroll
  for (int i = 0; i < 4; ++i) {
    float4 v = s[i];
    sum += v.x * v.x + v.y * v.y + v.z * v.z + v.w * v.w;
    t[4 * i + 0] = (__bf16)v.x;
    t[4 * i + 1] = (__bf16)v.y;
    t[4 * i + 2] = (__bf16)v.z;
    t[4 * i + 3] = (__bf16)v.w;
  }
  *reinterpret_cast<v16bf*>(dst + (size_t)row * DIMS + lane * 16) = t;

  if (c2 != nullptr) {
#pragma unroll
    for (int m = 16; m >= 1; m >>= 1) sum += __shfl_xor(sum, m, 32);
    if (lane == 0) c2[row] = sum;
  }
}

__global__ void k_zero(float* p) { p[0] = 0.0f; }

// ---------------------------------------------------------------------------
// Nearest-codebook search. Block = 256 threads (8 waves) = 2 M-tiles x
// 4 N-slices over 32 entity rows. dist = ||c||^2 - 2 * <x, c>  (x^2 constant).
// A block (32x512 bf16) staged once in LDS via async copy; A fragments are
// ds_load_b128 pairs, B fragments contiguous 32B row-major global loads.
// ---------------------------------------------------------------------------
__global__ void __launch_bounds__(256)
k_argmin(const __bf16* __restrict__ ent,
         const __bf16* __restrict__ cb,
         const float* __restrict__ c2,
         int* __restrict__ idx,
         int rows, int ncode) {
  __shared__ __align__(64) __bf16 Abuf[2][16][DIMS + APAD];   // ~33.8 KB
  __shared__ float s_v[2][16][4];
  __shared__ int   s_i[2][16][4];

  const int lane   = threadIdx.x & 31;
  const int wave   = threadIdx.x >> 5;
  const int mtile  = wave & 1;        // which 16-row tile of the 32-row block
  const int nslice = wave >> 1;       // which quarter of the codebook
  const int m0     = blockIdx.x * 32;

  // ---- Stage A: 32 rows x 1024B = 2048 16B chunks, 8 per thread (async). ---
#pragma unroll
  for (int it = 0; it < 8; ++it) {
    const int i     = threadIdx.x + it * 256;
    const int row   = i >> 6;          // 0..31
    const int chunk = i & 63;          // 16B chunk within the row
    int grow = m0 + row;
    if (grow >= rows) grow = rows - 1;
    async_copy16(&Abuf[row >> 4][row & 15][chunk * 8],
                 ent + (size_t)grow * DIMS + chunk * 8);
  }
  async_copy_join();

  const int khalf = (lane >> 4) * 16;             // K sub-span select
  const __bf16* aptr = &Abuf[mtile][lane & 15][khalf];
  const int ncol = lane & 15;                     // C/D: N = lane & 15

  float minv[8];
  int   mini[8];
#pragma unroll
  for (int r = 0; r < 8; ++r) { minv[r] = 3.4e38f; mini[r] = 0; }

  const int ntiles = ncode >> 4;                  // 128 tiles of 16 columns
  for (int nt = nslice; nt < ntiles; nt += 4) {
    const int n = nt * 16 + ncol;
    const __bf16* bptr = cb + (size_t)n * DIMS + khalf;
    v8f acc = {};
#pragma unroll
    for (int kt = 0; kt < DIMS / 32; ++kt) {      // 16 WMMA steps over K=512
      v16bf a = *reinterpret_cast<const v16bf*>(aptr + kt * 32);
      v16bf b = *reinterpret_cast<const v16bf*>(bptr + kt * 32);
      acc = __builtin_amdgcn_wmma_f32_16x16x32_bf16(
          false, a, false, b, (short)0, acc, false, false);
    }
    const float dbase = c2[n];
#pragma unroll
    for (int r = 0; r < 8; ++r) {
      float d = dbase - 2.0f * acc[r];            // M = r + 8*(lane>>4)
      if (d < minv[r]) { minv[r] = d; mini[r] = n; }
    }
  }

  // Reduce over the 16 lanes of each half (each half owns 8 M rows).
#pragma unroll
  for (int r = 0; r < 8; ++r) {
    float v = minv[r];
    int   i = mini[r];
#pragma unroll
    for (int m = 8; m >= 1; m >>= 1) {
      float ov = __shfl_xor(v, m, 32);
      int   oi = __shfl_xor(i, m, 32);
      if (ov < v) { v = ov; i = oi; }
    }
    if (lane == 0)  { s_v[mtile][r][nslice]     = v; s_i[mtile][r][nslice]     = i; }
    if (lane == 16) { s_v[mtile][r + 8][nslice] = v; s_i[mtile][r + 8][nslice] = i; }
  }
  __syncthreads();

  const int t = threadIdx.x;
  if (t < 32) {
    const int mt = t >> 4, row = t & 15;
    float bv = s_v[mt][row][0];
    int   bi = s_i[mt][row][0];
#pragma unroll
    for (int s = 1; s < 4; ++s)
      if (s_v[mt][row][s] < bv) { bv = s_v[mt][row][s]; bi = s_i[mt][row][s]; }
    const int m = m0 + mt * 16 + row;
    if (m < rows) idx[m] = bi;
  }
}

// ---------------------------------------------------------------------------
// vq_loss (fp32, gathered codebook rows) + idx emitted as float. Wave per row.
// ---------------------------------------------------------------------------
__global__ void k_loss(const float* __restrict__ ent,
                       const float* __restrict__ cbf,
                       const int* __restrict__ idx,
                       float* __restrict__ out_loss,
                       float* __restrict__ out_idx,
                       int rows, float scale) {
  const int row  = blockIdx.x * 8 + (threadIdx.x >> 5);
  const int lane = threadIdx.x & 31;
  if (row >= rows) return;
  const int k = idx[row];
  const float4* e = reinterpret_cast<const float4*>(ent + (size_t)row * DIMS) + lane * 4;
  const float4* c = reinterpret_cast<const float4*>(cbf + (size_t)k   * DIMS) + lane * 4;
  float sum = 0.0f;
#pragma unroll
  for (int i = 0; i < 4; ++i) {
    float4 ev = e[i], cv = c[i];
    float dx = cv.x - ev.x, dy = cv.y - ev.y, dz = cv.z - ev.z, dw = cv.w - ev.w;
    sum += dx * dx + dy * dy + dz * dz + dw * dw;
  }
#pragma unroll
  for (int m = 16; m >= 1; m >>= 1) sum += __shfl_xor(sum, m, 32);
  if (lane == 0) {
    atomicAdd(out_loss, sum * scale);
    out_idx[row] = (float)k;
  }
}

// ---------------------------------------------------------------------------
// score = query @ quant^T. Block = 64(M) x 128(N): 8 waves as 4x2, each wave a
// 16x64 tile via 4 accumulators (A fragment reused 4x). B rows gathered
// through idx from the L2-resident bf16 codebook (2 MB). Output stores are
// non-temporal (327 MB stream; keep L2 for codebook/query reuse).
// ---------------------------------------------------------------------------
__global__ void __launch_bounds__(256)
k_score(const __bf16* __restrict__ q,
        const __bf16* __restrict__ cb,
        const int* __restrict__ idx,
        float* __restrict__ out,
        int bdim) {
  const int lane = threadIdx.x & 31;
  const int wave = threadIdx.x >> 5;
  const int wm = wave >> 1, wn = wave & 1;
  const int m0 = blockIdx.y * 64 + wm * 16;
  const int n0 = blockIdx.x * 128 + wn * 64;
  const int khalf = (lane >> 4) * 16;

  const __bf16* aptr = q + (size_t)(m0 + (lane & 15)) * DIMS + khalf;

  int nidx[4];
  const __bf16* bptr[4];
#pragma unroll
  for (int j = 0; j < 4; ++j) {
    const int n = n0 + j * 16 + (lane & 15);
    nidx[j] = n;
    const int nc = (n < bdim) ? n : (bdim - 1);
    bptr[j] = cb + (size_t)idx[nc] * DIMS + khalf;
  }

  v8f acc0 = {}, acc1 = {}, acc2 = {}, acc3 = {};
#pragma unroll 4
  for (int kt = 0; kt < DIMS / 32; ++kt) {
    v16bf a = *reinterpret_cast<const v16bf*>(aptr + kt * 32);
    v16bf b0 = *reinterpret_cast<const v16bf*>(bptr[0] + kt * 32);
    acc0 = __builtin_amdgcn_wmma_f32_16x16x32_bf16(false, a, false, b0, (short)0, acc0, false, false);
    v16bf b1 = *reinterpret_cast<const v16bf*>(bptr[1] + kt * 32);
    acc1 = __builtin_amdgcn_wmma_f32_16x16x32_bf16(false, a, false, b1, (short)0, acc1, false, false);
    v16bf b2 = *reinterpret_cast<const v16bf*>(bptr[2] + kt * 32);
    acc2 = __builtin_amdgcn_wmma_f32_16x16x32_bf16(false, a, false, b2, (short)0, acc2, false, false);
    v16bf b3 = *reinterpret_cast<const v16bf*>(bptr[3] + kt * 32);
    acc3 = __builtin_amdgcn_wmma_f32_16x16x32_bf16(false, a, false, b3, (short)0, acc3, false, false);
  }

  const int mrow = m0 + 8 * (lane >> 4);          // C/D: M = r + 8*(lane>>4)
#pragma unroll
  for (int j = 0; j < 4; ++j) {
    if (nidx[j] >= bdim) continue;
    const v8f acc = (j == 0) ? acc0 : (j == 1) ? acc1 : (j == 2) ? acc2 : acc3;
#pragma unroll
    for (int r = 0; r < 8; ++r)
      __builtin_nontemporal_store(acc[r],
          &out[(size_t)(mrow + r) * bdim + nidx[j]]);
  }
}

// ---------------------------------------------------------------------------
// Host-side launcher.
// ---------------------------------------------------------------------------
extern "C" void kernel_launch(void* const* d_in, const int* in_sizes, int n_in,
                              void* d_out, int out_size, void* d_ws, size_t ws_size,
                              hipStream_t stream) {
  const float* query   = (const float*)d_in[0];   // (Q, 512)
  const float* entity  = (const float*)d_in[1];   // (B, 512)
  const float* codebook= (const float*)d_in[2];   // (K, 512)
  const int Q = in_sizes[0] / DIMS;               // 2048
  const int B = in_sizes[1] / DIMS;               // 40000
  const int K = in_sizes[2] / DIMS;               // 2048

  // Workspace partition.
  char* ws = (char*)d_ws;
  __bf16* cb_bf  = (__bf16*)ws;                  ws += (size_t)K * DIMS * sizeof(__bf16);
  __bf16* q_bf   = (__bf16*)ws;                  ws += (size_t)Q * DIMS * sizeof(__bf16);
  __bf16* ent_bf = (__bf16*)ws;                  ws += (size_t)B * DIMS * sizeof(__bf16);
  float*  c2     = (float*)ws;                   ws += (size_t)K * sizeof(float);
  int*    idx    = (int*)ws;

  float* out_score = (float*)d_out;                          // Q*B
  float* out_loss  = out_score + (size_t)Q * B;              // 1
  float* out_idx   = out_loss + 1;                           // B

  // 1. Conversions + codebook norms (+ zero the loss accumulator).
  k_zero<<<1, 1, 0, stream>>>(out_loss);
  k_convert<<<(K + 7) / 8, 256, 0, stream>>>(codebook, cb_bf, c2, K);
  k_convert<<<(Q + 7) / 8, 256, 0, stream>>>(query, q_bf, nullptr, Q);
  k_convert<<<(B + 7) / 8, 256, 0, stream>>>(entity, ent_bf, nullptr, B);

  // 2. Nearest-codebook argmin (bf16 WMMA distance GEMM).
  k_argmin<<<(B + 31) / 32, 256, 0, stream>>>(ent_bf, cb_bf, c2, idx, B, K);

  // 3. vq_loss (fp32) + idx output.
  const float scale = (1.0f + BETA) / ((float)B * (float)DIMS);
  k_loss<<<(B + 7) / 8, 256, 0, stream>>>(entity, codebook, idx, out_loss,
                                          out_idx, B, scale);

  // 4. Score GEMM with idx-gathered B fragments.
  dim3 grid((B + 127) / 128, Q / 64);
  k_score<<<grid, 256, 0, stream>>>(q_bf, cb_bf, idx, out_score, B);
}
